// SinkhornSim_5514738008420
// MI455X (gfx1250) — matrix operand order repaired and checked by reading, not compiled, not run on hardware
//
#include <hip/hip_runtime.h>
#include <hip/hip_bf16.h>

// Sinkhorn distance for B=16, C=64, N=2048 on gfx1250 (MI455X).
// Strategy:
//  1) softmax_swizzle: row softmax over C=64, emit xs/ys pre-swizzled into the
//     CDNA5 WMMA A-tile (16x32 f16) / B-tile (32x16 f16) per-lane layouts,
//     plus row squared-norms.
//  2) build_K: batched GEMM via v_wmma_f32_16x16x32_f16 (2 k-chunks of 32),
//     cost = max(x2+y2-2*dot, 0), K = exp(-cost/EPS), stored f16 (128 MiB,
//     fits MI455X's 192 MB L2 so the 100 Sinkhorn iterations stream from L2).
//  3) 100x { update_u (row matvec, v8h coalesced loads),
//            update_v (col matvec, v2h column-pair loads + LDS chunk reduce,
//            2048 waves to keep the bandwidth-bound phase fed) }.
//  4) distance: recompute cost with WMMA, accumulate u*K*v*cost via wave
//     reduction + atomicAdd; finalize 1 - d/(d+1e-9).

#define EPS_INV 1000.0f
#define NB 16
#define NN 2048
#define NC 64

typedef __attribute__((ext_vector_type(16))) _Float16 v16h;
typedef __attribute__((ext_vector_type(8)))  _Float16 v8h;
typedef __attribute__((ext_vector_type(2)))  _Float16 v2h;
typedef __attribute__((ext_vector_type(8)))  float    v8f;

__device__ __forceinline__ float wave_sum(float x) {
    #pragma unroll
    for (int off = 16; off > 0; off >>= 1) x += __shfl_xor(x, off, 32);
    return x;
}

// One wave (32 lanes) per row of 64 floats. is_b=0 -> A-tile swizzle (xs),
// is_b=1 -> B-tile swizzle (ys). Also writes sum of squares of softmaxed row.
__global__ void sinkhorn_softmax_swizzle(const float* __restrict__ in,
                                         _Float16* __restrict__ tiles,
                                         float* __restrict__ norms, int is_b) {
    int wave = threadIdx.x >> 5;
    int lane = threadIdx.x & 31;
    int row  = blockIdx.x * 8 + wave;          // row = b*2048 + n, 0..32767
    const float* rp = in + (size_t)row * NC;
    float a0 = rp[lane];
    float a1 = rp[lane + 32];
    float mx = fmaxf(a0, a1);
    #pragma unroll
    for (int off = 16; off > 0; off >>= 1) mx = fmaxf(mx, __shfl_xor(mx, off, 32));
    float e0 = __expf(a0 - mx), e1 = __expf(a1 - mx);
    float s = wave_sum(e0 + e1);
    float inv = 1.0f / s;
    float p0 = e0 * inv, p1 = e1 * inv;
    float sq = wave_sum(p0 * p0 + p1 * p1);
    if (lane == 0) norms[row] = sq;

    int b = row >> 11, n = row & 2047;
    int t16 = n >> 4, r16 = n & 15;
    #pragma unroll
    for (int e = 0; e < 2; ++e) {
        int c  = lane + e * 32;
        int kc = c >> 5, k = c & 31;
        int L, j;
        if (is_b) {                 // B-tile 32x16: lane = ((k>>3)&1)*16 + n, j = (k&7)+((k&16)?8:0)
            L = ((k & 8) ? 16 : 0) + r16;
            j = (k & 7) + ((k & 16) ? 8 : 0);
        } else {                    // A-tile 16x32: lane = M + ((k>>3)&1)*16, j = (k&7)+((k&16)?8:0)
            L = r16 + ((k & 8) ? 16 : 0);
            j = (k & 7) + ((k & 16) ? 8 : 0);
        }
        size_t idx = ((((size_t)b * 128 + t16) * 2 + kc) * 32 + L) * 16 + j;
        tiles[idx] = (_Float16)(e ? p1 : p0);
    }
}

__global__ void sinkhorn_init_uv(float* __restrict__ u, float* __restrict__ v,
                                 float* __restrict__ dist) {
    int i = blockIdx.x * blockDim.x + threadIdx.x;
    if (i < NB * NN) {
        u[i] = 1.0f / (float)NN;
        v[i] = 1.0f / (float)NN;
    }
    if (i < NB) dist[i] = 0.0f;
}

// One wave per 16x16 output tile: D = A(16x64) * B(64x16) via two k=32 WMMAs.
__global__ void sinkhorn_build_K(const _Float16* __restrict__ xsA,
                                 const _Float16* __restrict__ ysB,
                                 const float* __restrict__ x2,
                                 const float* __restrict__ y2,
                                 _Float16* __restrict__ K) {
    int lane = threadIdx.x & 31;
    int wid  = blockIdx.x * (blockDim.x >> 5) + (threadIdx.x >> 5);
    int b   = wid >> 14;                 // 128*128 tiles per batch
    int rem = wid & 16383;
    int nt  = rem >> 7, mt = rem & 127;

    const _Float16* Ab = xsA + ((((size_t)b * 128 + nt) * 2) * 32 + lane) * 16;
    const _Float16* Bb = ysB + ((((size_t)b * 128 + mt) * 2) * 32 + lane) * 16;
    v16h a0 = *(const v16h*)Ab;
    v16h a1 = *(const v16h*)(Ab + 512);  // kc=1 chunk: +32*16 halves
    v16h b0 = *(const v16h*)Bb;
    v16h b1 = *(const v16h*)(Bb + 512);

    v8f c = {};
    c = __builtin_amdgcn_wmma_f32_16x16x32_f16(false, a0, false, b0, (short)0, c, false, false);
    c = __builtin_amdgcn_wmma_f32_16x16x32_f16(false, a1, false, b1, (short)0, c, false, false);

    int hi = lane >> 4, nn = lane & 15;
    int n0 = nt * 16, m0 = mt * 16;
    float y2v = y2[b * NN + m0 + nn];
    _Float16* Kb = K + (size_t)b * NN * NN;
    #pragma unroll
    for (int r = 0; r < 8; ++r) {
        int M = r + hi * 8;
        float cost = fmaxf(x2[b * NN + n0 + M] + y2v - 2.0f * c[r], 0.0f);
        float kv = __expf(-cost * EPS_INV);
        Kb[(size_t)(n0 + M) * NN + (m0 + nn)] = (_Float16)kv;
    }
}

// u = 1 / (K v + 1e-8): one wave per row, 8x coalesced v8h loads per lane.
__global__ void sinkhorn_update_u(const _Float16* __restrict__ K,
                                  const float* __restrict__ v,
                                  float* __restrict__ u) {
    int lane = threadIdx.x & 31;
    int wid  = blockIdx.x * (blockDim.x >> 5) + (threadIdx.x >> 5); // b*2048+n
    const _Float16* Kr = K + (size_t)wid * NN;
    const float* vb = v + (wid >> 11) * NN;
    float acc = 0.0f;
    #pragma unroll
    for (int i = 0; i < 8; ++i) {
        int mb = (i * 32 + lane) * 8;
        v8h kv = *(const v8h*)(Kr + mb);
        v8f vv = *(const v8f*)(vb + mb);
        #pragma unroll
        for (int j = 0; j < 8; ++j) acc += (float)kv[j] * vv[j];
    }
    acc = wave_sum(acc);
    if (lane == 0) u[wid] = 1.0f / (acc + 1e-8f);
}

// v = 1 / (K^T u + 1e-8). Block = 256 threads: 64 column-pairs (v2h loads,
// 128 B per wave per row) x 4 row-chunks of 512, reduced through LDS.
// Grid = 16 batches * 16 column-groups = 256 blocks (2048 waves total).
__global__ void sinkhorn_update_v(const _Float16* __restrict__ K,
                                  const float* __restrict__ u,
                                  float* __restrict__ v) {
    __shared__ float smem[4][128];
    int t     = threadIdx.x;
    int cp    = t & 63;          // column pair within the 128-col group
    int chunk = t >> 6;          // 0..3, each covers 512 rows
    int b  = blockIdx.x >> 4;
    int m0 = (blockIdx.x & 15) * 128;
    const _Float16* Kb = K + (size_t)b * NN * NN + m0 + 2 * cp;
    const float* ub = u + b * NN;
    float a0 = 0.0f, a1 = 0.0f;
    int n0 = chunk * 512;
    #pragma unroll 4
    for (int n = n0; n < n0 + 512; ++n) {
        v2h kk = *(const v2h*)(Kb + (size_t)n * NN);
        float uu = ub[n];
        a0 += (float)kk[0] * uu;
        a1 += (float)kk[1] * uu;
    }
    smem[chunk][2 * cp]     = a0;
    smem[chunk][2 * cp + 1] = a1;
    __syncthreads();
    if (t < 128) {
        float s = smem[0][t] + smem[1][t] + smem[2][t] + smem[3][t];
        v[b * NN + m0 + t] = 1.0f / (s + 1e-8f);
    }
}

// Recompute cost/K per tile with WMMA, accumulate u*K*v*cost.
__global__ void sinkhorn_distance(const _Float16* __restrict__ xsA,
                                  const _Float16* __restrict__ ysB,
                                  const float* __restrict__ x2,
                                  const float* __restrict__ y2,
                                  const float* __restrict__ u,
                                  const float* __restrict__ v,
                                  float* __restrict__ dist) {
    int lane = threadIdx.x & 31;
    int wid  = blockIdx.x * (blockDim.x >> 5) + (threadIdx.x >> 5);
    int b   = wid >> 14;
    int rem = wid & 16383;
    int nt  = rem >> 7, mt = rem & 127;

    const _Float16* Ab = xsA + ((((size_t)b * 128 + nt) * 2) * 32 + lane) * 16;
    const _Float16* Bb = ysB + ((((size_t)b * 128 + mt) * 2) * 32 + lane) * 16;
    v16h a0 = *(const v16h*)Ab;
    v16h a1 = *(const v16h*)(Ab + 512);
    v16h b0 = *(const v16h*)Bb;
    v16h b1 = *(const v16h*)(Bb + 512);

    v8f c = {};
    c = __builtin_amdgcn_wmma_f32_16x16x32_f16(false, a0, false, b0, (short)0, c, false, false);
    c = __builtin_amdgcn_wmma_f32_16x16x32_f16(false, a1, false, b1, (short)0, c, false, false);

    int hi = lane >> 4, nn = lane & 15;
    int n0 = nt * 16, m0 = mt * 16;
    float y2v = y2[b * NN + m0 + nn];
    float vv  = v[b * NN + m0 + nn];
    float part = 0.0f;
    #pragma unroll
    for (int r = 0; r < 8; ++r) {
        int M = r + hi * 8;
        float cost = fmaxf(x2[b * NN + n0 + M] + y2v - 2.0f * c[r], 0.0f);
        float kv = __expf(-cost * EPS_INV);
        part += u[b * NN + n0 + M] * kv * vv * cost;
    }
    part = wave_sum(part);
    if (lane == 0) atomicAdd(&dist[b], part);
}

__global__ void sinkhorn_finalize(const float* __restrict__ dist,
                                  float* __restrict__ out) {
    int b = threadIdx.x;
    if (b < NB) {
        float d = dist[b];
        out[b] = 1.0f - d / (d + 1e-9f);
    }
}

extern "C" void kernel_launch(void* const* d_in, const int* in_sizes, int n_in,
                              void* d_out, int out_size, void* d_ws, size_t ws_size,
                              hipStream_t stream) {
    const float* x = (const float*)d_in[0];
    const float* y = (const float*)d_in[1];
    float* out = (float*)d_out;

    // Workspace layout (bytes):
    //   xsA:  16*128*2*32*16 f16 = 4 MiB
    //   ysB:  4 MiB
    //   x2,y2,u,v: 32768 f32 each = 128 KiB each
    //   dist: 64 f32 (padding to 256B)
    //   K:    16*2048*2048 f16 = 128 MiB   (total ~143 MiB)
    char* ws = (char*)d_ws;
    _Float16* xsA = (_Float16*)ws;
    _Float16* ysB = xsA + (size_t)2097152;
    float* x2   = (float*)(ysB + (size_t)2097152);
    float* y2   = x2 + 32768;
    float* u    = y2 + 32768;
    float* v    = u + 32768;
    float* dist = v + 32768;
    _Float16* K = (_Float16*)(dist + 64);

    sinkhorn_softmax_swizzle<<<4096, 256, 0, stream>>>(x, xsA, x2, 0);
    sinkhorn_softmax_swizzle<<<4096, 256, 0, stream>>>(y, ysB, y2, 1);
    sinkhorn_init_uv<<<128, 256, 0, stream>>>(u, v, dist);
    sinkhorn_build_K<<<32768, 256, 0, stream>>>(xsA, ysB, x2, y2, K);

    for (int it = 0; it < 100; ++it) {
        sinkhorn_update_u<<<4096, 256, 0, stream>>>(K, v, u);
        sinkhorn_update_v<<<256, 256, 0, stream>>>(K, u, v);
    }

    sinkhorn_distance<<<32768, 256, 0, stream>>>(xsA, ysB, x2, y2, u, v, dist);
    sinkhorn_finalize<<<1, 16, 0, stream>>>(dist, out);
}
